// Model_5360119185494
// MI455X (gfx1250) — compile-verified, matrix operand order checked by reading, not run on hardware
//
#include <hip/hip_runtime.h>
#include <hip/hip_bf16.h>
#include <math.h>

// ---------------------------------------------------------------------------
// Two-layer LSTM + GELU + FC, fused persistent kernel for gfx1250 (MI455X).
//  - bf16 WMMA (v_wmma_f32_16x16x32_bf16) for all recurrent GEMMs, f32 state.
//  - One persistent kernel runs the whole 256-step scan; grid-wide spin
//    barriers separate GEMM phases from element-wise gate phases.
//  - Layer-2 input projection fused into its recurrence via concatenated
//    A = [gelu(h1_t) | h2_t] (K = 768).
//  - Weights pair-packed to bf16 once; B fragments are single coalesced
//    dword loads matching the CDNA5 16-bit A/B VGPR layout.
//  - Per-wave register tile is MPW x NPW (4x2 / 2x2): square-ish blocking
//    minimizes fragment loads per wmma (48 dwords per 8 wmma vs 72 for 1x8).
//  - Gate pre-activations stored TILE-MAJOR: G[(tile*8 + r)*32 + lane]; each
//    D-tile store is 8 immediate-offset coalesced b32 stores off ONE base.
// ---------------------------------------------------------------------------

#define Bsz 512
#define Ssz 256
#define Hsz 512
#define H2sz 256
#define Psz 21
#define G1sz (4 * Hsz)    // 2048
#define G2sz (4 * H2sz)   // 1024
#define KA2 (Hsz + H2sz)  // 768
#define NT1 (G1sz / 16)   // 128 n-tiles, layer 1
#define NT2 (G2sz / 16)   // 64 n-tiles, layer 2

#define NBLK 64
#define TPB 256
#define NWAVES ((NBLK * TPB) / 32)  // 512

typedef __attribute__((ext_vector_type(16))) __bf16 v16bf;
typedef __attribute__((ext_vector_type(8)))  float  v8f;

union BF16Frag {
  v16bf    v;
  unsigned u[8];
};

__device__ __forceinline__ unsigned short f2bf(float f) {
  unsigned u = __float_as_uint(f);
  u += 0x7FFFu + ((u >> 16) & 1u);  // round-to-nearest-even
  return (unsigned short)(u >> 16);
}

__device__ __forceinline__ float sigmoidf(float x) {
  return 1.0f / (1.0f + expf(-x));
}

// Grid-wide generation barrier (persistent kernel; all blocks resident).
__device__ __forceinline__ void grid_sync(unsigned* bar, unsigned nblk) {
  __syncthreads();
  if (threadIdx.x == 0) {
    volatile unsigned* vbar = bar;
    __threadfence();
    unsigned gen = vbar[1];
    if (atomicAdd(&bar[0], 1u) == nblk - 1u) {
      vbar[0] = 0;
      __threadfence();
      atomicAdd(&bar[1], 1u);
    } else {
      while (vbar[1] == gen) { __builtin_amdgcn_s_sleep(2); }
    }
    __threadfence();
  }
  __syncthreads();
}

// One timestep GEMM phase: Gs(tile-major) = A[M,K](bf16,row-major) x
// Wp[K,N](bf16 pair-packed: dword = {bf16 k, bf16 k+1} at Wp[(k>>1)*N + n]).
// Each wave owns an MPW x NPW block of 16x16 output tiles; A and B fragments
// are each reused NPW / MPW times respectively.
// Output: Gs[((mt*ntn + nt)*8 + r)*32 + lane] = D[mt*16 + 8*(lane>>4) + r,
//                                                 nt*16 + (lane&15)]
template <int MPW, int NPW>
__device__ __forceinline__ void gemm_step(const unsigned short* __restrict__ A,
                                          int lda,
                                          const unsigned* __restrict__ Wp,
                                          int N, int K,
                                          float* __restrict__ Gs,
                                          int waveId) {
  const int lane = threadIdx.x & 31;
  const int ln = lane & 15;     // N (B,C,D) / M (A) index within tile
  const int lh = lane >> 4;     // half-wave select (K split)
  const int ntn = N >> 4;
  const int nGroups = ntn / NPW;
  const int nt0 = (waveId % nGroups) * NPW;
  const int mt0 = (waveId / nGroups) * MPW;

  v8f acc[MPW][NPW];
#pragma unroll
  for (int m = 0; m < MPW; ++m)
#pragma unroll
    for (int n = 0; n < NPW; ++n)
#pragma unroll
      for (int r = 0; r < 8; ++r) acc[m][n][r] = 0.0f;

  for (int kk = 0; kk < K; kk += 32) {
    // B fragments (32x16 bf16 each): lane ln = column, K striped per ISA.
    BF16Frag bf[NPW];
#pragma unroll
    for (int n = 0; n < NPW; ++n)
#pragma unroll
      for (int r = 0; r < 8; ++r) {
        const int ko = 16 * (r >> 2) + 8 * lh + 2 * (r & 3);
        bf[n].u[r] = Wp[(size_t)((kk + ko) >> 1) * N + (nt0 + n) * 16 + ln];
      }
    // Prefetch next K-slab of the packed weights (speculative, L2 hint).
    __builtin_prefetch(&Wp[(size_t)((kk + 32) >> 1) * N + nt0 * 16 + ln], 0, 1);

#pragma unroll
    for (int m = 0; m < MPW; ++m) {
      const unsigned* Arow =
          (const unsigned*)(A + (size_t)((mt0 + m) * 16 + ln) * lda);
      BF16Frag af;
#pragma unroll
      for (int r = 0; r < 8; ++r) {
        const int ko = 16 * (r >> 2) + 8 * lh + 2 * (r & 3);
        af.u[r] = Arow[(kk + ko) >> 1];  // pair (k,k+1) contiguous row-major
      }
#pragma unroll
      for (int n = 0; n < NPW; ++n)
        acc[m][n] = __builtin_amdgcn_wmma_f32_16x16x32_bf16(
            false, af.v, false, bf[n].v, (short)0, acc[m][n], false, false);
    }
  }

  // Tile-major store: one base address per tile, 8 immediate-offset stores,
  // each coalesced across the wave (32 lanes x 4B contiguous).
#pragma unroll
  for (int m = 0; m < MPW; ++m)
#pragma unroll
    for (int n = 0; n < NPW; ++n) {
      const int tileId = (mt0 + m) * ntn + nt0 + n;
      float* gbase = Gs + (size_t)tileId * 256 + lane;
#pragma unroll
      for (int r = 0; r < 8; ++r) gbase[r * 32] = acc[m][n][r];
    }
}

// Read one gate pre-activation from tile-major Gs for logical (row b, col g).
__device__ __forceinline__ float gate_read(const float* __restrict__ Gs,
                                           int ntn, int b, int g) {
  const int tile = (b >> 4) * ntn + (g >> 4);
  const int r = b & 7;
  const int lane = (g & 15) + ((b >> 3) & 1) * 16;
  return Gs[(size_t)(tile * 8 + r) * 32 + lane];
}

// ---------------------------------------------------------------------------
// Setup kernels
// ---------------------------------------------------------------------------

// W_hh1 [2048,512] f32 -> pair-packed bf16 [256][2048] dwords (B = W_hh1^T).
__global__ void pack_w1(const float* __restrict__ W, unsigned* __restrict__ Wp) {
  int idx = blockIdx.x * blockDim.x + threadIdx.x;
  if (idx >= (Hsz / 2) * G1sz) return;
  int kp = idx / G1sz;
  int n  = idx - kp * G1sz;
  unsigned lo = f2bf(W[(size_t)n * Hsz + 2 * kp]);
  unsigned hi = f2bf(W[(size_t)n * Hsz + 2 * kp + 1]);
  Wp[idx] = lo | (hi << 16);
}

// [W_ih2 | W_hh2]^T concat along K (768) -> pair-packed [384][1024] dwords.
__global__ void pack_w2(const float* __restrict__ Wih,
                        const float* __restrict__ Whh,
                        unsigned* __restrict__ Wp) {
  int idx = blockIdx.x * blockDim.x + threadIdx.x;
  if (idx >= (KA2 / 2) * G2sz) return;
  int kp = idx / G2sz;
  int n  = idx - kp * G2sz;
  int k  = 2 * kp;
  unsigned lo, hi;
  if (k < Hsz) {
    lo = f2bf(Wih[(size_t)n * Hsz + k]);
    hi = f2bf(Wih[(size_t)n * Hsz + k + 1]);
  } else {
    lo = f2bf(Whh[(size_t)n * H2sz + (k - Hsz)]);
    hi = f2bf(Whh[(size_t)n * H2sz + (k - Hsz) + 1]);
  }
  Wp[idx] = lo | (hi << 16);
}

// Zero-init recurrent state + barrier (h0 = c0 = 0 per reference).
__global__ void init_state(unsigned short* __restrict__ h1,
                           float* __restrict__ c1,
                           unsigned short* __restrict__ A2,
                           float* __restrict__ c2,
                           unsigned* __restrict__ bar) {
  int i = blockIdx.x * blockDim.x + threadIdx.x;
  int stride = gridDim.x * blockDim.x;
  for (int k = i; k < Bsz * Hsz; k += stride) h1[k] = 0;
  for (int k = i; k < Bsz * Hsz; k += stride) c1[k] = 0.0f;
  for (int k = i; k < Bsz * KA2; k += stride) A2[k] = 0;
  for (int k = i; k < Bsz * H2sz; k += stride) c2[k] = 0.0f;
  if (i < 2) bar[i] = 0;
}

// ---------------------------------------------------------------------------
// Persistent fused 2-layer LSTM scan
// ---------------------------------------------------------------------------
__global__ __launch_bounds__(TPB) void lstm_fused(
    const float* __restrict__ x,      // [B,S] (input size 1 squeezed)
    const float* __restrict__ W_ih1,  // [4H]
    const float* __restrict__ b_ih1, const float* __restrict__ b_hh1,
    const float* __restrict__ b_ih2, const float* __restrict__ b_hh2,
    const unsigned* __restrict__ W1p, const unsigned* __restrict__ W2p,
    unsigned short* __restrict__ h1,  // bf16 [B,H]
    float* __restrict__ c1,           // [B,H]
    float* __restrict__ gates1,       // tile-major, B*4H floats
    unsigned short* __restrict__ A2,  // bf16 [B,768] = [gelu(h1) | h2]
    float* __restrict__ c2,           // [B,H2]
    float* __restrict__ gates2,       // tile-major, B*4H2 floats
    float* __restrict__ h2f,          // [B,H2] f32 (final step survives)
    unsigned* __restrict__ bar) {
  const int tid = blockIdx.x * blockDim.x + threadIdx.x;
  const int nThreads = gridDim.x * blockDim.x;
  const int waveId = tid >> 5;

  for (int t = 0; t < Ssz; ++t) {
    // ---- Phase A1: gates1 = h1 @ W_hh1^T  (M=512, N=2048, K=512) ----------
    // 4096 tiles / 512 waves = 4x2 (MxN) register tile per wave.
    gemm_step<4, 2>(h1, Hsz, W1p, G1sz, Hsz, gates1, waveId);
    grid_sync(bar, gridDim.x);

    // ---- Phase B1: layer-1 gate math + GELU, write bf16 h1 and A2[:, :H] --
    for (int e = tid; e < Bsz * Hsz; e += nThreads) {
      const int b = e >> 9;          // /H
      const int j = e & (Hsz - 1);
      const float xv = x[(size_t)b * Ssz + t];
      const float pi = gate_read(gates1, NT1, b, j)
                       + xv * W_ih1[j]           + b_ih1[j]           + b_hh1[j];
      const float pf = gate_read(gates1, NT1, b, Hsz + j)
                       + xv * W_ih1[Hsz + j]     + b_ih1[Hsz + j]     + b_hh1[Hsz + j];
      const float pg = gate_read(gates1, NT1, b, 2 * Hsz + j)
                       + xv * W_ih1[2 * Hsz + j] + b_ih1[2 * Hsz + j] + b_hh1[2 * Hsz + j];
      const float po = gate_read(gates1, NT1, b, 3 * Hsz + j)
                       + xv * W_ih1[3 * Hsz + j] + b_ih1[3 * Hsz + j] + b_hh1[3 * Hsz + j];
      const float c = sigmoidf(pf) * c1[e] + sigmoidf(pi) * tanhf(pg);
      c1[e] = c;
      const float h = sigmoidf(po) * tanhf(c);
      h1[e] = f2bf(h);
      const float a = 0.5f * h * (1.0f + erff(h * 0.70710678118654752440f));
      A2[(size_t)b * KA2 + j] = f2bf(a);  // layer-2 input (gelu)
    }
    grid_sync(bar, gridDim.x);

    // ---- Phase A2: gates2 = [a_t|h2] @ [W_ih2|W_hh2]^T (M=512,N=1024,K=768)
    // 2048 tiles / 512 waves = 2x2 (MxN) register tile per wave.
    gemm_step<2, 2>(A2, KA2, W2p, G2sz, KA2, gates2, waveId);
    grid_sync(bar, gridDim.x);

    // ---- Phase B2: layer-2 gate math, write bf16 h2 into A2[:, H:] --------
    for (int e = tid; e < Bsz * H2sz; e += nThreads) {
      const int b = e >> 8;          // /H2
      const int j = e & (H2sz - 1);
      const float pi = gate_read(gates2, NT2, b, j)            + b_ih2[j]            + b_hh2[j];
      const float pf = gate_read(gates2, NT2, b, H2sz + j)     + b_ih2[H2sz + j]     + b_hh2[H2sz + j];
      const float pg = gate_read(gates2, NT2, b, 2 * H2sz + j) + b_ih2[2 * H2sz + j] + b_hh2[2 * H2sz + j];
      const float po = gate_read(gates2, NT2, b, 3 * H2sz + j) + b_ih2[3 * H2sz + j] + b_hh2[3 * H2sz + j];
      const float c = sigmoidf(pf) * c2[e] + sigmoidf(pi) * tanhf(pg);
      c2[e] = c;
      const float h = sigmoidf(po) * tanhf(c);
      A2[(size_t)b * KA2 + Hsz + j] = f2bf(h);
      h2f[e] = h;  // final step's value feeds the FC head
    }
    grid_sync(bar, gridDim.x);
  }
}

// ---------------------------------------------------------------------------
// Final FC head: out[b,p] = h2_last[b,:] . fc_w[p,:] + fc_b[p]   (tiny)
// ---------------------------------------------------------------------------
__global__ void fc_head(const float* __restrict__ h2f,
                        const float* __restrict__ fc_w,
                        const float* __restrict__ fc_b,
                        float* __restrict__ out) {
  int idx = blockIdx.x * blockDim.x + threadIdx.x;
  if (idx >= Bsz * Psz) return;
  int b = idx / Psz;
  int p = idx - b * Psz;
  float acc = fc_b[p];
#pragma unroll 4
  for (int j = 0; j < H2sz; ++j)
    acc += h2f[(size_t)b * H2sz + j] * fc_w[(size_t)p * H2sz + j];
  out[idx] = acc;
}

// ---------------------------------------------------------------------------
extern "C" void kernel_launch(void* const* d_in, const int* in_sizes, int n_in,
                              void* d_out, int out_size, void* d_ws,
                              size_t ws_size, hipStream_t stream) {
  const float* x     = (const float*)d_in[0];   // [B,S,1]
  const float* W_ih1 = (const float*)d_in[1];   // [4H,1]
  const float* W_hh1 = (const float*)d_in[2];   // [4H,H]
  const float* b_ih1 = (const float*)d_in[3];
  const float* b_hh1 = (const float*)d_in[4];
  const float* W_ih2 = (const float*)d_in[5];   // [4H2,H]
  const float* W_hh2 = (const float*)d_in[6];   // [4H2,H2]
  const float* b_ih2 = (const float*)d_in[7];
  const float* b_hh2 = (const float*)d_in[8];
  const float* fc_w  = (const float*)d_in[9];   // [P,H2]
  const float* fc_b  = (const float*)d_in[10];
  float* out = (float*)d_out;

  // Workspace carve-up (~12.5 MB total)
  char* ws = (char*)d_ws;
  size_t off = 0;
  auto carve = [&](size_t bytes) -> void* {
    off = (off + 255) & ~(size_t)255;
    void* p = ws + off;
    off += bytes;
    return p;
  };
  unsigned*       bar    = (unsigned*)carve(256);
  unsigned*       W1p    = (unsigned*)carve((size_t)(Hsz / 2) * G1sz * 4);   // 2 MB
  unsigned*       W2p    = (unsigned*)carve((size_t)(KA2 / 2) * G2sz * 4);   // 1.5 MB
  unsigned short* h1     = (unsigned short*)carve((size_t)Bsz * Hsz * 2);    // 512 KB
  float*          c1     = (float*)carve((size_t)Bsz * Hsz * 4);             // 1 MB
  float*          gates1 = (float*)carve((size_t)Bsz * G1sz * 4);            // 4 MB
  unsigned short* A2     = (unsigned short*)carve((size_t)Bsz * KA2 * 2);    // 768 KB
  float*          c2     = (float*)carve((size_t)Bsz * H2sz * 4);            // 512 KB
  float*          gates2 = (float*)carve((size_t)Bsz * G2sz * 4);            // 2 MB
  float*          h2f    = (float*)carve((size_t)Bsz * H2sz * 4);            // 512 KB
  (void)ws_size; (void)in_sizes; (void)n_in; (void)out_size;

  // One-time (per launch) weight packing + state init.
  pack_w1<<<((Hsz / 2) * G1sz + TPB - 1) / TPB, TPB, 0, stream>>>(W_hh1, W1p);
  pack_w2<<<((KA2 / 2) * G2sz + TPB - 1) / TPB, TPB, 0, stream>>>(W_ih2, W_hh2, W2p);
  init_state<<<256, TPB, 0, stream>>>(h1, c1, A2, c2, bar);

  // Persistent fused scan over all 256 timesteps.
  lstm_fused<<<NBLK, TPB, 0, stream>>>(x, W_ih1, b_ih1, b_hh1, b_ih2, b_hh2,
                                       W1p, W2p, h1, c1, gates1, A2, c2,
                                       gates2, h2f, bar);

  // FC head on the last hidden state.
  fc_head<<<(Bsz * Psz + TPB - 1) / TPB, TPB, 0, stream>>>(h2f, fc_w, fc_b, out);
}
